// MambaBlock_36945308680368
// MI455X (gfx1250) — matrix operand order repaired and checked by reading, not compiled, not run on hardware
//
#include <hip/hip_runtime.h>
#include <math.h>

typedef __attribute__((ext_vector_type(16))) _Float16 v16h;
typedef __attribute__((ext_vector_type(8)))  _Float16 v8h;
typedef __attribute__((ext_vector_type(4)))  _Float16 v4h;
typedef __attribute__((ext_vector_type(8)))  float    v8f;

#define Bdim   2
#define Ldim   2048
#define Ddim   768
#define DIdim  1536
#define DSdim  16
#define DTRdim 48
#define Kconv  4
#define ROWS   (Bdim*Ldim)   // 4096
#define DBCLD  128           // padded row stride of dbc (was 80)

__device__ __forceinline__ float sigmoidf_(float x){ return 1.f/(1.f+__expf(-x)); }

// ---------------- f32 -> f16 conversion (4-wide) ----------------
__global__ void cvt_f32_f16(const float* __restrict__ in, _Float16* __restrict__ out, int n4){
  int i = blockIdx.x*blockDim.x + threadIdx.x;
  if (i < n4){
    float4 v = ((const float4*)in)[i];
    v4h o = {(_Float16)v.x, (_Float16)v.y, (_Float16)v.z, (_Float16)v.w};
    ((v4h*)out)[i] = o;
  }
}

// ---------------- f32 -> f16 with zero padding to (Rout x Cout) ----------------
__global__ void cvt_pad_f16(const float* __restrict__ in, _Float16* __restrict__ out,
                            int Rout, int Cout, int Rin, int Cin){
  int i = blockIdx.x*blockDim.x + threadIdx.x;
  if (i >= Rout*Cout) return;
  int r = i / Cout, c = i % Cout;
  out[i] = (r < Rin && c < Cin) ? (_Float16)in[(size_t)r*Cin + c] : (_Float16)0.f;
}

// ---------------- layernorm (row = one block) ----------------
__global__ void layernorm_f16(const float* __restrict__ x, const float* __restrict__ g,
                              const float* __restrict__ b, _Float16* __restrict__ out, int D){
  __shared__ float s1[256], s2[256];
  int row = blockIdx.x, tid = threadIdx.x;
  const float* xr = x + (size_t)row*D;
  float ls = 0.f, lq = 0.f;
  for (int i = tid; i < D; i += 256){ float v = xr[i]; ls += v; lq += v*v; }
  s1[tid] = ls; s2[tid] = lq; __syncthreads();
  for (int o = 128; o > 0; o >>= 1){
    if (tid < o){ s1[tid] += s1[tid+o]; s2[tid] += s2[tid+o]; }
    __syncthreads();
  }
  float mu = s1[0] / D;
  float var = s2[0] / D - mu*mu;
  float rs = rsqrtf(var + 1e-5f);
  for (int i = tid; i < D; i += 256)
    out[(size_t)row*D + i] = (_Float16)((xr[i]-mu)*rs*g[i] + b[i]);
}

// ---------------- WMMA GEMM (fully tile-aligned: M%64==0, N%64==0, K%32==0) ----------------
// C = act(A @ W^T + bias) (+add).  A: MxK f16 (lda); W: NxK f16 (ldw).
#define BM  64
#define BN  64
#define BKt 32
#define LDT 40   // padded LDS row stride (f16): 80B rows, 16B-aligned, conflict-free

// Async 16B global->LDS copy (CDNA5 GLOBAL_LOAD_ASYNC_TO_LDS_B128, ASYNCcnt-tracked).
__device__ __forceinline__
void stage_tile_async(_Float16* __restrict__ dst, const _Float16* __restrict__ src,
                      int ld, int row0, int k0, int lr, int lc){
  const _Float16* sp = src + (size_t)(row0 + lr)*ld + k0 + lc;
  unsigned lds = (unsigned)(size_t)(dst + lr*LDT + lc);   // low 32b = LDS byte offset
  asm volatile("global_load_async_to_lds_b128 %0, %1, off"
               :: "v"(lds), "v"((unsigned long long)(size_t)sp)
               : "memory");
}
__device__ __forceinline__ void async_wait0(){
  asm volatile("s_wait_asynccnt 0x0" ::: "memory");
}

__global__ __launch_bounds__(256)
void gemm_wmma_f16(const _Float16* __restrict__ A, int lda,
                   const _Float16* __restrict__ W, int ldw,
                   const float* __restrict__ bias,
                   const float* __restrict__ add,          // MxN (ld=N) or null
                   float* __restrict__ Cf, int ldcf,
                   _Float16* __restrict__ Ch, int ldch, int coloff,
                   int N, int K, int act)
{
  __shared__ _Float16 As[2][BM*LDT];
  __shared__ _Float16 Ws[2][BN*LDT];
  const int tid = threadIdx.x;
  const int rowBase = blockIdx.y * BM;
  const int colBase = blockIdx.x * BN;
  const int wid = tid >> 5, lane = tid & 31;
  const int wm = wid & 3;                 // wave M-tile (0..3)
  const int wnBase = (wid >> 2) * 32;     // wave N strip: two 16-wide tiles
  const int hl = lane >> 4, l15 = lane & 15;
  const int lr = tid >> 2;                // staging: tile row 0..63
  const int lc = (tid & 3) * 8;           // staging: k-chunk 0/8/16/24

  v8f acc0 = {}, acc1 = {};

  const int nk = K / BKt;
  stage_tile_async(As[0], A, lda, rowBase, 0, lr, lc);
  stage_tile_async(Ws[0], W, ldw, colBase, 0, lr, lc);
  async_wait0();
  __syncthreads();

  for (int it = 0; it < nk; ++it){
    const int cur = it & 1;
    if (it + 1 < nk){                     // async-stage next tiles into alternate buffer
      stage_tile_async(As[cur^1], A, lda, rowBase, (it+1)*BKt, lr, lc);
      stage_tile_async(Ws[cur^1], W, ldw, colBase, (it+1)*BKt, lr, lc);
    }

    // A fragment (16x32): row = l15, K = hl*16 + e (16 contiguous)
    const _Float16* arow = &As[cur][(wm*16 + l15)*LDT + hl*16];
    v8h alo = *(const v8h*)(arow);
    v8h ahi = *(const v8h*)(arow + 8);
    v16h a;
    #pragma unroll
    for (int i = 0; i < 8; i++){ a[i] = alo[i]; a[i+8] = ahi[i]; }

    // B fragments (32x16): N = l15, K = hl*16 + e (16 contiguous)
    const _Float16* w0 = &Ws[cur][(wnBase      + l15)*LDT + hl*16];
    const _Float16* w1 = &Ws[cur][(wnBase + 16 + l15)*LDT + hl*16];
    v8h b0lo = *(const v8h*)(w0), b0hi = *(const v8h*)(w0 + 8);
    v8h b1lo = *(const v8h*)(w1), b1hi = *(const v8h*)(w1 + 8);
    v16h b0, b1;
    #pragma unroll
    for (int i = 0; i < 8; i++){ b0[i]=b0lo[i]; b0[i+8]=b0hi[i]; b1[i]=b1lo[i]; b1[i+8]=b1hi[i]; }

    acc0 = __builtin_amdgcn_wmma_f32_16x16x32_f16(false, a, false, b0, (short)0, acc0, false, false);
    acc1 = __builtin_amdgcn_wmma_f32_16x16x32_f16(false, a, false, b1, (short)0, acc1, false, false);

    if (it + 1 < nk) async_wait0();       // next tiles landed (ran under the WMMAs)
    __syncthreads();                      // publish to all waves; one barrier per K-step
  }

  // epilogue: lane<16 -> (M=r, N=l15); lane>=16 -> (M=r+8, N=l15)
  #pragma unroll
  for (int r = 0; r < 8; r++){
    int m = rowBase + wm*16 + hl*8 + r;
    #pragma unroll
    for (int q = 0; q < 2; q++){
      int n = colBase + wnBase + q*16 + l15;
      float v = (q == 0) ? acc0[r] : acc1[r];
      if (bias) v += bias[n];
      if (act == 1) v = (v > 20.f) ? v : log1pf(__expf(v));
      else if (act == 2){
        float u = 0.7978845608f*(v + 0.044715f*v*v*v);
        v = 0.5f*v*(1.f + tanhf(u));
      }
      if (add) v += add[(size_t)m*N + n];
      if (Cf) Cf[(size_t)m*ldcf + n] = v;
      if (Ch) Ch[(size_t)m*ldch + coloff + n] = (_Float16)v;
    }
  }
}

// ---------------- causal / anti-causal depthwise conv (K=4) + SiLU ----------------
__global__ void conv_silu(const float* __restrict__ xz, const float* __restrict__ cw,
                          const float* __restrict__ cb, float* __restrict__ of,
                          _Float16* __restrict__ oh, int dir){
  int idx = blockIdx.x*blockDim.x + threadIdx.x;
  int d = idx % DIdim;
  int t = (idx / DIdim) % Ldim;
  int b = idx / (DIdim*Ldim);
  float acc = cb[d];
  #pragma unroll
  for (int k = 0; k < Kconv; k++){
    int tt = dir ? (t + (Kconv-1) - k) : (t - (Kconv-1) + k);
    if (tt >= 0 && tt < Ldim)
      acc += cw[d*Kconv + k] * xz[((size_t)b*Ldim + tt)*(2*DIdim) + d];
  }
  float s = acc * sigmoidf_(acc);
  size_t o = ((size_t)b*Ldim + t)*DIdim + d;
  of[o] = s; oh[o] = (_Float16)s;
}

// ---------------- selective scan (4 lanes per channel, 4 states each) ----------------
__global__ void ssm_scan(const float* __restrict__ dt, const float* __restrict__ dbc,
                         const float* __restrict__ xcs, const float* __restrict__ xz,
                         const float* __restrict__ A_log, const float* __restrict__ Dp,
                         _Float16* __restrict__ y, int dir){
  int tid = threadIdx.x;
  int lane = tid & 31;
  int sg = lane & 3;                               // state group (4 states)
  int ch = (blockIdx.x*256 + tid) >> 2;            // 0 .. B*DI-1
  int d = ch % DIdim, b = ch / DIdim;
  float a[4], h[4] = {0.f,0.f,0.f,0.f};
  #pragma unroll
  for (int s = 0; s < 4; s++) a[s] = -__expf(A_log[d*DSdim + sg*4 + s]);
  float Dd = Dp[d];
  for (int step = 0; step < Ldim; step++){
    int t = dir ? (Ldim-1-step) : step;
    size_t r = (size_t)b*Ldim + t;
    float dtv = dt[r*DIdim + d];
    float xv  = xcs[r*DIdim + d];
    const float* bc = dbc + r*DBCLD + DTRdim + sg*4;  // B slice; C at +16
    float dx = dtv*xv, yv = 0.f;
    #pragma unroll
    for (int s = 0; s < 4; s++){
      h[s] = __expf(dtv*a[s])*h[s] + dx*bc[s];
      yv += h[s]*bc[DSdim + s];
    }
    yv += __shfl_xor(yv, 1, 32);
    yv += __shfl_xor(yv, 2, 32);
    if (sg == 0){
      float zv = xz[r*(2*DIdim) + DIdim + d];
      float g = yv + xv*Dd;
      y[r*DIdim + d] = (_Float16)(g * (zv * sigmoidf_(zv)));
    }
  }
}

// ---------------- host orchestration ----------------
extern "C" void kernel_launch(void* const* d_in, const int* in_sizes, int n_in,
                              void* d_out, int out_size, void* d_ws, size_t ws_size,
                              hipStream_t stream){
  const float* x        = (const float*)d_in[0];
  const float* norm_g   = (const float*)d_in[1];
  const float* norm_b   = (const float*)d_in[2];
  const float* mix_w    = (const float*)d_in[21];
  const float* mix_b    = (const float*)d_in[22];
  const float* ffn_w1   = (const float*)d_in[23];
  const float* ffn_b1   = (const float*)d_in[24];
  const float* ffn_w2   = (const float*)d_in[25];
  const float* ffn_b2   = (const float*)d_in[26];
  const float* ffn_ng   = (const float*)d_in[27];
  const float* ffn_nb   = (const float*)d_in[28];

  char* p = (char*)d_ws;
  auto alloc = [&](size_t bytes)->char*{ char* r = p; p += (bytes + 255) & ~(size_t)255; return r; };

  _Float16* xn_h   = (_Float16*)alloc((size_t)ROWS*Ddim*2);
  float*    xz_f   = (float*)   alloc((size_t)ROWS*2*DIdim*4);
  float*    xcs_f  = (float*)   alloc((size_t)ROWS*DIdim*4);
  _Float16* xcs_h  = (_Float16*)alloc((size_t)ROWS*DIdim*2);
  float*    dbc_f  = (float*)   alloc((size_t)ROWS*DBCLD*4);   // padded stride 128
  _Float16* dbc_h  = (_Float16*)alloc((size_t)ROWS*DBCLD*2);
  float*    dt_f   = (float*)   alloc((size_t)ROWS*DIdim*4);
  _Float16* y_h    = (_Float16*)alloc((size_t)ROWS*DIdim*2);
  _Float16* cat_h  = (_Float16*)alloc((size_t)ROWS*DIdim*2);
  float*    h_f    = (float*)   alloc((size_t)ROWS*Ddim*4);
  _Float16* hn_h   = (_Float16*)alloc((size_t)ROWS*Ddim*2);
  _Float16* g_h    = (_Float16*)alloc((size_t)ROWS*4*Ddim*2);
  _Float16* win_h  = (_Float16*)alloc((size_t)2*DIdim*Ddim*2);
  _Float16* wxp_h  = (_Float16*)alloc((size_t)DBCLD*DIdim*2);  // padded to 128 rows
  _Float16* wdt_h  = (_Float16*)alloc((size_t)DIdim*64*2);     // padded to 64 cols
  _Float16* wout_h = (_Float16*)alloc((size_t)Ddim*DIdim*2);
  _Float16* wmix_h = (_Float16*)alloc((size_t)Ddim*2*Ddim*2);
  _Float16* w1_h   = (_Float16*)alloc((size_t)4*Ddim*Ddim*2);
  _Float16* w2_h   = (_Float16*)alloc((size_t)Ddim*4*Ddim*2);

  auto cvt = [&](const float* src, _Float16* dst, int n){
    cvt_f32_f16<<<(n/4 + 255)/256, 256, 0, stream>>>(src, dst, n/4);
  };
  auto cvtpad = [&](const float* src, _Float16* dst, int Rout, int Cout, int Rin, int Cin){
    int n = Rout*Cout;
    cvt_pad_f16<<<(n + 255)/256, 256, 0, stream>>>(src, dst, Rout, Cout, Rin, Cin);
  };
  auto gemm = [&](const _Float16* A, int lda, const _Float16* W, int ldw,
                  const float* bias, const float* add,
                  float* Cf, int ldcf, _Float16* Ch, int ldch, int coloff,
                  int M, int N, int K, int act){
    dim3 grid(N/BN, M/BM);
    gemm_wmma_f16<<<grid, 256, 0, stream>>>(A, lda, W, ldw, bias, add,
                                            Cf, ldcf, Ch, ldch, coloff, N, K, act);
  };

  // shared weights (once per call)
  cvt(mix_w,  wmix_h, Ddim*2*Ddim);
  cvt(ffn_w1, w1_h,   4*Ddim*Ddim);
  cvt(ffn_w2, w2_h,   Ddim*4*Ddim);

  // LN1
  layernorm_f16<<<ROWS, 256, 0, stream>>>(x, norm_g, norm_b, xn_h, Ddim);

  for (int dir = 0; dir < 2; dir++){
    int o = dir ? 12 : 3;
    const float* in_w    = (const float*)d_in[o+0];
    const float* conv_w  = (const float*)d_in[o+1];
    const float* conv_b  = (const float*)d_in[o+2];
    const float* xproj_w = (const float*)d_in[o+3];
    const float* dt_w    = (const float*)d_in[o+4];
    const float* dt_b    = (const float*)d_in[o+5];
    const float* A_log   = (const float*)d_in[o+6];
    const float* Dp      = (const float*)d_in[o+7];
    const float* out_w   = (const float*)d_in[o+8];

    cvt(in_w,  win_h,  2*DIdim*Ddim);
    cvt(out_w, wout_h, Ddim*DIdim);
    cvtpad(xproj_w, wxp_h, DBCLD, DIdim, 80, DIdim);     // 80x1536 -> 128x1536 (zero rows)
    cvtpad(dt_w,    wdt_h, DIdim, 64,    DIdim, DTRdim); // 1536x48 -> 1536x64 (zero cols)

    // xz = xn @ in_w^T   (4096 x 3072 x 768)
    gemm(xn_h, Ddim, win_h, Ddim, nullptr, nullptr,
         xz_f, 2*DIdim, nullptr, 0, 0, ROWS, 2*DIdim, Ddim, 0);

    // depthwise conv (causal fwd / anti-causal bwd) + SiLU
    conv_silu<<<(ROWS*DIdim)/256, 256, 0, stream>>>(xz_f, conv_w, conv_b, xcs_f, xcs_h, dir);

    // dbc = xc @ xproj_pad^T  (4096 x 128 x 1536); cols 80..127 are zeros
    gemm(xcs_h, DIdim, wxp_h, DIdim, nullptr, nullptr,
         dbc_f, DBCLD, dbc_h, DBCLD, 0, ROWS, DBCLD, DIdim, 0);

    // dt = softplus(dbc[:, :64] @ dt_w_pad^T + dt_b)  (4096 x 1536 x 64, lda=128)
    // dt_w cols 48..63 are zero, so dbc's B-values in cols 48..63 contribute 0.
    gemm(dbc_h, DBCLD, wdt_h, 64, dt_b, nullptr,
         dt_f, DIdim, nullptr, 0, 0, ROWS, DIdim, 64, 1);

    // selective scan (reverse time for dir==1), fused gate, f16 out
    dim3 sg((Bdim*DIdim*4)/256);
    ssm_scan<<<sg, 256, 0, stream>>>(dt_f, dbc_f, xcs_f, xz_f, A_log, Dp, y_h, dir);

    // out-proj into concat buffer at column offset dir*768 (4096 x 768 x 1536)
    gemm(y_h, DIdim, wout_h, DIdim, nullptr, nullptr,
         nullptr, 0, cat_h, 2*Ddim, dir*Ddim, ROWS, Ddim, DIdim, 0);
  }

  // h = concat @ mix_w^T + mix_b + residual
  gemm(cat_h, 2*Ddim, wmix_h, 2*Ddim, mix_b, x,
       h_f, Ddim, nullptr, 0, 0, ROWS, Ddim, 2*Ddim, 0);

  // LN2
  layernorm_f16<<<ROWS, 256, 0, stream>>>(h_f, ffn_ng, ffn_nb, hn_h, Ddim);

  // FFN1: gelu(hn @ w1^T + b1) -> f16
  gemm(hn_h, Ddim, w1_h, Ddim, ffn_b1, nullptr,
       nullptr, 0, g_h, 4*Ddim, 0, ROWS, 4*Ddim, Ddim, 2);

  // FFN2: g @ w2^T + b2 + h -> d_out (f32)
  gemm(g_h, 4*Ddim, w2_h, 4*Ddim, ffn_b2, h_f,
       (float*)d_out, Ddim, nullptr, 0, 0, ROWS, Ddim, 4*Ddim, 0);
}
